// Self_Attention_58110907514947
// MI455X (gfx1250) — matrix operand order repaired and checked by reading, not compiled, not run on hardware
//
#include <hip/hip_runtime.h>

// ---------------------------------------------------------------------------
// Channel self-attention for MI455X (gfx1250, wave32, WMMA).
// All GEMMs use v_wmma_f32_16x16x32_f16 (f16 in, f32 accumulate).
// Roofline: ~22.5 GFLOP vs ~460 MB HBM traffic -> memory-bound (~20us floor
// at 23.3 TB/s); f16 WMMA compute (~10us) stays under the floor.
// This revision: (a) x streamed exactly once in the QKV GEMM (M=192 in one
// block tile), (b) v streamed exactly once in the output GEMM (M=512 tile),
// (c) all global staging uses 16-byte vector loads instead of u16/b32.
// ---------------------------------------------------------------------------

typedef __attribute__((ext_vector_type(16))) _Float16 v16h;
typedef __attribute__((ext_vector_type(8)))  _Float16 v8h;
typedef __attribute__((ext_vector_type(8)))  float    v8f;

#define BDIM 16
#define CDIM 512
#define CPD  64
#define NDIM 4096
#define QKV_ROWS 192   // q(64) | k(64) | v(64)

// ---- WMMA fragment helpers (CDNA5 ISA 7.12.2 layouts, wave32) -------------

__device__ __forceinline__ v16h make_frag(v8h lo, v8h hi) {
  return __builtin_shufflevector(lo, hi, 0,1,2,3,4,5,6,7,8,9,10,11,12,13,14,15);
}

// A fragment, 16x32 (MxK), row-major source with row stride ld (elements).
// lane<16: row=lane,    K = {0..7, 16..23}
// lane>=16: row=lane-16, K = {8..15, 24..31}
__device__ __forceinline__ v16h load_a(const _Float16* src, int ld) {
  const int lane = threadIdx.x & 31;
  const int r  = lane & 15;
  const int kq = (lane >> 4) << 3;            // 0 or 8
  const _Float16* p = src + r * ld + kq;
  v8h lo = *(const v8h*)(p);
  v8h hi = *(const v8h*)(p + 16);
  return make_frag(lo, hi);
}

// B fragment, 32x16 (KxN), from an N-major (transposed) tile Bt[n][k] with
// row stride ld. lane<16: col=lane, K={0..7,16..23} ; lane>=16: col=lane-16, K={8..15,24..31}
__device__ __forceinline__ v16h load_b(const _Float16* bt, int ld) {
  const int lane = threadIdx.x & 31;
  const int n  = lane & 15;
  const int k0 = (lane >> 4) << 3;            // 0 or 8
  const _Float16* p = bt + n * ld + k0;
  v8h lo = *(const v8h*)(p);
  v8h hi = *(const v8h*)(p + 16);
  return make_frag(lo, hi);
}

__device__ __forceinline__ v8f wmma16(v16h a, v16h b, v8f c) {
  return __builtin_amdgcn_wmma_f32_16x16x32_f16(false, a, false, b,
                                                (short)0, c, false, false);
}

// ---------------------------------------------------------------------------
// Kernel 0: pack wq|wk|wv -> f16 [192][512], biases -> f32 [192]
// ---------------------------------------------------------------------------
__global__ void prep_weights(const float* __restrict__ wq, const float* __restrict__ bq,
                             const float* __restrict__ wk, const float* __restrict__ bk,
                             const float* __restrict__ wv, const float* __restrict__ bv,
                             _Float16* __restrict__ wqkv, float* __restrict__ bqkv) {
  int idx = blockIdx.x * blockDim.x + threadIdx.x;
  if (idx < QKV_ROWS * CDIM) {
    int o = idx / CDIM, c = idx % CDIM;
    const float* w = (o < CPD) ? wq : (o < 2 * CPD ? wk : wv);
    wqkv[idx] = (_Float16)w[(o & 63) * CDIM + c];
  }
  if (idx < QKV_ROWS) {
    const float* bs = (idx < CPD) ? bq : (idx < 2 * CPD ? bk : bv);
    bqkv[idx] = bs[idx & 63];
  }
}

// ---------------------------------------------------------------------------
// Kernel 1: qkv[b][o][n] = wqkv @ x[b] + bias   (192x512)x(512x4096) per batch
// Block tile: M=192 (ALL output rows -> x streamed exactly once), N=64,
// K-step=32. 8 waves = 4(M-slices of 3 subtiles) x 2(N); wave = 6 subtiles.
// x converted f32->f16 (float4 loads) while staging transposed into LDS.
// ---------------------------------------------------------------------------
#define K1_KT  32
#define K1_NT  64
#define K1_LDA 40
#define K1_LDB 40

__global__ __launch_bounds__(256) void qkv_gemm(
    const float* __restrict__ x, const _Float16* __restrict__ wqkv,
    const float* __restrict__ bqkv, _Float16* __restrict__ qkv) {
  __shared__ __attribute__((aligned(16))) _Float16 sA[QKV_ROWS * K1_LDA]; // 192x32
  __shared__ __attribute__((aligned(16))) _Float16 sB[K1_NT * K1_LDB];    // 64x32 (x^T)
  const int b   = blockIdx.z;
  const int nt  = blockIdx.x * K1_NT;
  const int tid = threadIdx.x;
  const int w   = tid >> 5;
  const int wm  = w & 3;      // 4 M-slices; wave owns rows {wm*16, 64+wm*16, 128+wm*16}
  const int wn  = w >> 2;     // 2 N-groups (32 cols each)

  const float* xb = x + (size_t)b * CDIM * NDIM;
  v8f acc[3][2] = {};

  for (int kt = 0; kt < CDIM; kt += K1_KT) {
    __syncthreads();
    // stage weights 192x32: 768 v8h chunks, 3 per thread (16B loads)
    for (int i = tid; i < QKV_ROWS * K1_KT / 8; i += 256) {
      int m  = i >> 2;
      int c8 = (i & 3) << 3;
      *(v8h*)&sA[m * K1_LDA + c8] = *(const v8h*)&wqkv[m * CDIM + kt + c8];
    }
    // stage x^T 64x32: 512 float4 chunks, 2 per thread, cvt f32->f16
    for (int i = tid; i < K1_KT * K1_NT / 4; i += 256) {
      int k = i >> 4;            // 0..31
      int n = (i & 15) << 2;     // 0..60 step 4
      float4 f = *(const float4*)&xb[(size_t)(kt + k) * NDIM + nt + n];
      sB[(n + 0) * K1_LDB + k] = (_Float16)f.x;
      sB[(n + 1) * K1_LDB + k] = (_Float16)f.y;
      sB[(n + 2) * K1_LDB + k] = (_Float16)f.z;
      sB[(n + 3) * K1_LDB + k] = (_Float16)f.w;
    }
    __syncthreads();
    v16h a[3];
    for (int mi = 0; mi < 3; ++mi)
      a[mi] = load_a(sA + (mi * 64 + wm * 16) * K1_LDA, K1_LDA);
    for (int s = 0; s < 2; ++s) {
      v16h bf = load_b(sB + (wn * 32 + s * 16) * K1_LDB, K1_LDB);
      for (int mi = 0; mi < 3; ++mi)
        acc[mi][s] = wmma16(a[mi], bf, acc[mi][s]);
    }
  }

  const int lane = tid & 31;
  const int col  = lane & 15;
  const int rb   = (lane >> 4) << 3;
  _Float16* outb = qkv + (size_t)b * QKV_ROWS * NDIM;
  for (int mi = 0; mi < 3; ++mi) {
    for (int s = 0; s < 2; ++s) {
      int n = nt + wn * 32 + s * 16 + col;
      for (int g = 0; g < 8; ++g) {
        int o = mi * 64 + wm * 16 + rb + g;
        outb[(size_t)o * NDIM + n] = (_Float16)(acc[mi][s][g] + bqkv[o]);
      }
    }
  }
}

// ---------------------------------------------------------------------------
// Kernel 2a: partial energy[i][j] = sum_n k[i,n] q[j,n] over an n-chunk.
// B-fragment wants Bt[j][n] == q's natural layout -> both fragments load
// straight from global (16B loads), no LDS. 8 n-chunks x 16 batches.
// ---------------------------------------------------------------------------
__global__ __launch_bounds__(256) void energy_partial(
    const _Float16* __restrict__ qkv, float* __restrict__ epart) {
  const int nchunk = blockIdx.x;                 // 8 chunks of 512
  const int b      = blockIdx.y;
  const int tid    = threadIdx.x;
  const int w      = tid >> 5;
  const _Float16* qb = qkv + (size_t)b * QKV_ROWS * NDIM;
  const _Float16* kb = qb + (size_t)CPD * NDIM;
  const int si  = (2 * w) >> 2;                  // wave -> 2 adjacent subtiles
  const int sj0 = (2 * w) & 3, sj1 = sj0 + 1;

  v8f acc0 = {}, acc1 = {};
  const int n0 = nchunk * 512;
  const _Float16* aB = kb + (size_t)si  * 16 * NDIM + n0;
  const _Float16* b0 = qb + (size_t)sj0 * 16 * NDIM + n0;
  const _Float16* b1 = qb + (size_t)sj1 * 16 * NDIM + n0;
  for (int n = 0; n < 512; n += 32) {
    v16h a = load_a(aB + n, NDIM);
    acc0 = wmma16(a, load_b(b0 + n, NDIM), acc0);
    acc1 = wmma16(a, load_b(b1 + n, NDIM), acc1);
  }
  const int lane = tid & 31, col = lane & 15, rb = (lane >> 4) << 3;
  float* ep = epart + ((size_t)b * 8 + nchunk) * (CPD * CPD);
  for (int g = 0; g < 8; ++g) {
    ep[(si * 16 + rb + g) * CPD + sj0 * 16 + col] = acc0[g];
    ep[(si * 16 + rb + g) * CPD + sj1 * 16 + col] = acc1[g];
  }
}

// ---------------------------------------------------------------------------
// Kernel 2b: reduce partials -> softmax rows -> fold output projection:
// m = wo @ attn  (512x64 per batch; kills the (B,C',N) intermediate).
// ---------------------------------------------------------------------------
__global__ __launch_bounds__(256) void softmax_m(
    const float* __restrict__ epart, const float* __restrict__ wo,
    _Float16* __restrict__ mOut) {
  __shared__ float sE[CPD * 65];
  const int b   = blockIdx.x;
  const int tid = threadIdx.x;
  const float* ep = epart + (size_t)b * 8 * CPD * CPD;
  for (int idx = tid; idx < CPD * CPD; idx += 256) {
    float s = 0.f;
    for (int p = 0; p < 8; ++p) s += ep[p * CPD * CPD + idx];
    sE[(idx >> 6) * 65 + (idx & 63)] = s;
  }
  __syncthreads();
  if (tid < CPD) {
    float* row = sE + tid * 65;
    float mx = row[0];
    for (int j = 1; j < CPD; ++j) mx = fmaxf(mx, row[j]);
    float sum = 0.f;
    for (int j = 0; j < CPD; ++j) { float e = __expf(row[j] - mx); row[j] = e; sum += e; }
    float inv = 1.f / sum;
    for (int j = 0; j < CPD; ++j) row[j] *= inv;
  }
  __syncthreads();
  _Float16* mb = mOut + (size_t)b * CDIM * CPD;
  for (int idx = tid; idx < CDIM * CPD; idx += 256) {
    int c = idx >> 6, j = idx & 63;
    const float* wrow = wo + c * CPD;
    float sum = 0.f;
    for (int i = 0; i < CPD; ++i) sum += wrow[i] * sE[i * 65 + j];
    mb[idx] = (_Float16)sum;
  }
}

// ---------------------------------------------------------------------------
// Kernel 3: out = gamma*(m @ v + bo) + x   ((512x64)x(64x4096) per batch)
// Block tile: M=512 (ALL channels -> v streamed exactly once), N=64, K=64.
// 8 waves; wave = 64 rows x 64 cols = 16 subtiles, 32 WMMAs. m is 1 MB total
// and L2-resident. Fused residual epilogue.
// ---------------------------------------------------------------------------
#define K3_NT  64
#define K3_LDV 72

__global__ __launch_bounds__(256) void out_gemm(
    const _Float16* __restrict__ qkv, const _Float16* __restrict__ mIn,
    const float* __restrict__ bo, const float* __restrict__ gamma,
    const float* __restrict__ x, float* __restrict__ out) {
  __shared__ __attribute__((aligned(16))) _Float16 sV[K3_NT * K3_LDV]; // v^T 64x64
  const int b   = blockIdx.z;
  const int nt  = blockIdx.x * K3_NT;
  const int tid = threadIdx.x;
  const int w   = tid >> 5;

  const _Float16* vb = qkv + (size_t)b * QKV_ROWS * NDIM + (size_t)2 * CPD * NDIM;
  // stage v^T: 512 v8h chunks (16B loads), 2 per thread -> sV[n][j]
  for (int i = tid; i < CPD * K3_NT / 8; i += 256) {
    int j  = i >> 3;           // 0..63
    int n8 = (i & 7) << 3;     // 0..56
    v8h vv = *(const v8h*)&vb[(size_t)j * NDIM + nt + n8];
    for (int e = 0; e < 8; ++e) sV[(n8 + e) * K3_LDV + j] = vv[e];
  }
  __syncthreads();

  const _Float16* mb = mIn + (size_t)b * CDIM * CPD + (size_t)w * 64 * CPD;
  v8f acc[4][4] = {};
  for (int k = 0; k < CPD; k += 32) {
    v16h a[4];
    for (int mi = 0; mi < 4; ++mi)
      a[mi] = load_a(mb + (size_t)mi * 16 * CPD + k, CPD);
    for (int s = 0; s < 4; ++s) {
      v16h f = load_b(sV + s * 16 * K3_LDV + k, K3_LDV);
      for (int mi = 0; mi < 4; ++mi)
        acc[mi][s] = wmma16(a[mi], f, acc[mi][s]);
    }
  }

  const int lane = tid & 31, col = lane & 15, rb = (lane >> 4) << 3;
  const float g0 = gamma[0];
  const float* xb = x + (size_t)b * CDIM * NDIM;
  float* ob = out + (size_t)b * CDIM * NDIM;
  for (int mi = 0; mi < 4; ++mi) {
    for (int s = 0; s < 4; ++s) {
      int n = nt + s * 16 + col;
      for (int gg = 0; gg < 8; ++gg) {
        int c = w * 64 + mi * 16 + rb + gg;
        size_t off = (size_t)c * NDIM + n;
        ob[off] = g0 * (acc[mi][s][gg] + bo[c]) + xb[off];
      }
    }
  }
}

// ---------------------------------------------------------------------------
// Workspace layout (bytes):
//   wqkv f16 [192][512]          @ 0         (196608)
//   bqkv f32 [192]               @ 196608    (1024, padded)
//   qkv  f16 [16][192][4096]     @ 197632    (25165824)
//   m    f16 [16][512][64]       @ 25363456  (1048576)
//   epart f32 [16][8][64][64]    @ 26412032  (2097152)   total ~28.5 MB
// ---------------------------------------------------------------------------
extern "C" void kernel_launch(void* const* d_in, const int* in_sizes, int n_in,
                              void* d_out, int out_size, void* d_ws, size_t ws_size,
                              hipStream_t stream) {
  const float* x     = (const float*)d_in[0];
  const float* wq    = (const float*)d_in[1];
  const float* bq    = (const float*)d_in[2];
  const float* wk    = (const float*)d_in[3];
  const float* bk    = (const float*)d_in[4];
  const float* wv    = (const float*)d_in[5];
  const float* bv    = (const float*)d_in[6];
  const float* wo    = (const float*)d_in[7];
  const float* bo    = (const float*)d_in[8];
  const float* gamma = (const float*)d_in[9];
  float* out = (float*)d_out;

  char* ws = (char*)d_ws;
  _Float16* wqkv  = (_Float16*)(ws);
  float*    bqkv  = (float*)   (ws + 196608);
  _Float16* qkv   = (_Float16*)(ws + 197632);
  _Float16* m     = (_Float16*)(ws + 25363456);
  float*    epart = (float*)   (ws + 26412032);

  prep_weights<<<(QKV_ROWS * CDIM + 255) / 256, 256, 0, stream>>>(
      wq, bq, wk, bk, wv, bv, wqkv, bqkv);
  qkv_gemm<<<dim3(NDIM / K1_NT, 1, BDIM), 256, 0, stream>>>(x, wqkv, bqkv, qkv);
  energy_partial<<<dim3(8, BDIM), 256, 0, stream>>>(qkv, epart);
  softmax_m<<<BDIM, 256, 0, stream>>>(epart, wo, m);
  out_gemm<<<dim3(NDIM / K3_NT, 1, BDIM), 256, 0, stream>>>(
      qkv, m, bo, gamma, x, out);
}
